// Attention_20272245637793
// MI455X (gfx1250) — matrix operand order repaired and checked by reading, not compiled, not run on hardware
//
#include <hip/hip_runtime.h>
#include <math.h>

typedef __bf16 bf16;
typedef __attribute__((ext_vector_type(16))) __bf16 v16bf;
typedef __attribute__((ext_vector_type(8)))  __bf16 v8bf;
typedef __attribute__((ext_vector_type(4)))  __bf16 v4bf;
typedef __attribute__((ext_vector_type(8)))  float  v8f;
typedef __attribute__((ext_vector_type(4)))  float  v4f;
typedef __attribute__((ext_vector_type(4)))  unsigned int v4u;
typedef __attribute__((ext_vector_type(4)))  int    v4i;
typedef __attribute__((ext_vector_type(8)))  int    v8i;

#define B_   2
#define S_   2048
#define H_   2048
#define NH_  16
#define NKV_ 8
#define GQ_  (NH_ / NKV_)
#define D_   128
#define M_   (B_ * S_)
#define KVW_ (NKV_ * D_)   /* 1024 */

union FragAB { v16bf v; v8bf h[2]; };

__device__ __forceinline__ v8f wmma_bf16f32(v16bf a, v16bf b, v8f c) {
  // D = A(16x32 bf16) x B(32x16 bf16) + C(16x16 f32)
  return __builtin_amdgcn_wmma_f32_16x16x32_bf16(false, a, false, b,
                                                 (short)0, c, false, false);
}

// ---------------------------------------------------------------- TDM 2-D tile
// Issue a TENSOR_LOAD_TO_LDS for a 2-D bf16 tile:
//   tile_d0 elems/row, tile_d1 rows, row stride = stride0 elems,
//   LDS rows padded by (pad_amount_code+1) DWORDs every 2^(pad_iv_code+1) DWORDs.
// D# bit layout per CDNA5 ISA.
__device__ __forceinline__ void tdm_load_2d(const bf16* gptr, unsigned lds_addr,
                                            int tile_d0, int tile_d1,
                                            int stride0,
                                            int pad_iv_code, int pad_amt_code) {
  unsigned long long ga = (unsigned long long)(size_t)gptr;
  v4u g0;
  g0.x = 1u;                                          // count=1, user mode
  g0.y = lds_addr;                                    // lds_addr[31:0]
  g0.z = (unsigned)ga;                                // global_addr[31:0]
  g0.w = (unsigned)((ga >> 32) & 0x01FFFFFFu) | (2u << 30);  // addr[56:32], type=2
  v8i g1;
  g1[0] = (int)((1u << 16)                            // data_size = 2 bytes
              | (1u << 20)                            // pad_enable
              | ((unsigned)pad_iv_code << 22)
              | ((unsigned)pad_amt_code << 25));
  g1[1] = (int)(((unsigned)tile_d0 & 0xFFFFu) << 16);               // tensor_dim0 lo
  g1[2] = (int)((((unsigned)tile_d0 >> 16) & 0xFFFFu)
              | (((unsigned)tile_d1 & 0xFFFFu) << 16));             // tdim0 hi | tdim1 lo
  g1[3] = (int)((((unsigned)tile_d1 >> 16) & 0xFFFFu)
              | ((unsigned)tile_d0 << 16));                         // tdim1 hi | tile_dim0
  g1[4] = (int)((unsigned)tile_d1 & 0xFFFFu);                       // tile_dim1 (tile_dim2=0)
  g1[5] = stride0;                                                  // tensor_dim0_stride lo32
  g1[6] = 0;                                                        // stride0 hi | stride1 lo
  g1[7] = 0;
  v4i z4 = {0, 0, 0, 0};
#if defined(__clang_major__) && __clang_major__ >= 23
  v8i z8 = {0, 0, 0, 0, 0, 0, 0, 0};
  __builtin_amdgcn_tensor_load_to_lds(g0, g1, z4, z4, z8, 0);
#else
  __builtin_amdgcn_tensor_load_to_lds(g0, g1, z4, z4, 0);
#endif
}

// ---------------------------------------------------------------- fp32 -> bf16
__global__ void cvt_f32_to_bf16(const float* __restrict__ src,
                                bf16* __restrict__ dst, int n4) {
  int i = blockIdx.x * blockDim.x + threadIdx.x;
  int stride = gridDim.x * blockDim.x;
  const v4f* s4 = (const v4f*)src;
  v4bf* d4 = (v4bf*)dst;
  for (; i < n4; i += stride) {
    v4f v = s4[i];
    v4bf o;
    o.x = (bf16)v.x; o.y = (bf16)v.y; o.z = (bf16)v.z; o.w = (bf16)v.w;
    d4[i] = o;
  }
}

// --------------------------------------------- C[M,N] = A[M,K] * W[N,K]^T (bf16)
// block: 256 thr (8 waves, 2x4), tile 64(M) x 128(N) x 32(K)
// Tiles staged by the Tensor Data Mover, double buffered (TENSORcnt pipelining).
template <typename OutT>
__global__ void __launch_bounds__(256)
gemm_xwT(const bf16* __restrict__ A, const bf16* __restrict__ W,
         OutT* __restrict__ C, int M, int N, int K) {
  __shared__ alignas(16) bf16 sA[2][64][40];    // 32->40 pad done by TDM
  __shared__ alignas(16) bf16 sB[2][128][40];

  const int t    = threadIdx.x;
  const int wv   = t >> 5;
  const int lane = t & 31;
  const int hl   = lane >> 4;   // lane half
  const int lm   = lane & 15;
  const int wm   = wv >> 2;     // 0..1 -> 32 rows each
  const int wn   = wv & 3;      // 0..3 -> 32 cols each
  const int m0   = blockIdx.y * 64;
  const int n0   = blockIdx.x * 128;

  v8f acc[2][2] = {};
  const int nIter = K >> 5;

  if (wv == 0) {   // prologue: DMA first K-tile into buffer 0
    tdm_load_2d(&A[(size_t)m0 * K], (unsigned)(size_t)&sA[0][0][0], 32, 64, K, 3, 3);
    tdm_load_2d(&W[(size_t)n0 * K], (unsigned)(size_t)&sB[0][0][0], 32, 128, K, 3, 3);
  }

  for (int it = 0; it < nIter; ++it) {
    const int buf = it & 1;
    if (wv == 0) {
      if (it + 1 < nIter) {     // DMA next tile into other buffer, then wait
        const int kc = (it + 1) << 5;
        tdm_load_2d(&A[(size_t)m0 * K + kc],
                    (unsigned)(size_t)&sA[buf ^ 1][0][0], 32, 64, K, 3, 3);
        tdm_load_2d(&W[(size_t)n0 * K + kc],
                    (unsigned)(size_t)&sB[buf ^ 1][0][0], 32, 128, K, 3, 3);
        __builtin_amdgcn_s_wait_tensorcnt(2);   // current tile (in-order) landed
      } else {
        __builtin_amdgcn_s_wait_tensorcnt(0);
      }
    }
    __syncthreads();

    FragAB fa[2], fb[2];
#pragma unroll
    for (int mt = 0; mt < 2; ++mt) {       // A layout: K {0-7,16-23}/{8-15,24-31}
      const int r = wm * 32 + mt * 16 + lm;
      fa[mt].h[0] = *(const v8bf*)&sA[buf][r][hl * 8];
      fa[mt].h[1] = *(const v8bf*)&sA[buf][r][hl * 8 + 16];
    }
#pragma unroll
    for (int nt = 0; nt < 2; ++nt) {       // B layout: K {0-7,16-23}/{8-15,24-31}
      const int r = wn * 32 + nt * 16 + lm;
      fb[nt].h[0] = *(const v8bf*)&sB[buf][r][hl * 8];
      fb[nt].h[1] = *(const v8bf*)&sB[buf][r][hl * 8 + 16];
    }
#pragma unroll
    for (int mt = 0; mt < 2; ++mt)
#pragma unroll
      for (int nt = 0; nt < 2; ++nt)
        acc[mt][nt] = wmma_bf16f32(fa[mt].v, fb[nt].v, acc[mt][nt]);

    __syncthreads();   // all reads done before this buffer is DMA-overwritten
  }

#pragma unroll
  for (int mt = 0; mt < 2; ++mt)
#pragma unroll
    for (int nt = 0; nt < 2; ++nt)
#pragma unroll
      for (int r = 0; r < 8; ++r) {
        int row = m0 + wm * 32 + mt * 16 + hl * 8 + r;
        int col = n0 + wn * 32 + nt * 16 + lm;
        C[(size_t)row * N + col] = (OutT)acc[mt][nt][r];
      }
}

// ------------------------------------------------------------------ RoPE inplace
__global__ void rope_inplace(bf16* __restrict__ X, int nheads, int rowlen,
                             int total) {
  int idx = blockIdx.x * blockDim.x + threadIdx.x;
  if (idx >= total) return;
  int j   = idx & 63;                      // D/2 = 64
  int h   = (idx >> 6) % nheads;
  int row = idx / (64 * nheads);
  int pos = row & (S_ - 1);
  // inv_freq = 10000^(-2j/D) = exp(-ln(1e4) * j / 64)
  float inv_freq = __expf(-9.210340371976184f * (float)j * (1.0f / 64.0f));
  float ang = (float)pos * inv_freq;
  float s, c;
  __sincosf(ang, &s, &c);
  bf16* p = X + (size_t)row * rowlen + h * D_ + j;
  float x1 = (float)p[0];
  float x2 = (float)p[64];
  p[0]  = (bf16)(x1 * c - x2 * s);
  p[64] = (bf16)(x2 * c + x1 * s);
}

// --------------------------------------------------------------- flash attention
// block = 256 thr (8 waves); block tile = 128 q rows of one head; wave = 16 rows
// K tile staged by TDM (padded rows in hardware); V staged transposed by VALU.
__global__ void __launch_bounds__(256)
flash_attn(const bf16* __restrict__ Q, const bf16* __restrict__ Kg,
           const bf16* __restrict__ Vg, bf16* __restrict__ O) {
  __shared__ alignas(16) bf16 sK[64][136];     // K tile, row-major [k][d] (TDM pad)
  __shared__ alignas(16) bf16 sVT[128][72];    // V tile transposed [d][k]
  __shared__ alignas(16) bf16 sP[8][16][72];   // per-wave P scratch [q][k]

  const int t    = threadIdx.x;
  const int wv   = t >> 5;
  const int lane = t & 31;
  const int hl   = lane >> 4;
  const int lm   = lane & 15;

  const int qt  = blockIdx.x;
  const int h   = blockIdx.y;
  const int b   = blockIdx.z;
  const int kvo = (h / GQ_) * D_;
  const int q0  = qt * 128 + wv * 16;      // wave's first q row (within seq)

  // Q fragments (A layout), kept in registers for the whole block
  FragAB qf[4];
  {
    const bf16* qrow = Q + (size_t)(b * S_ + q0 + lm) * H_ + h * D_;
#pragma unroll
    for (int dc = 0; dc < 4; ++dc) {
      qf[dc].h[0] = *(const v8bf*)(qrow + dc * 32 + hl * 8);
      qf[dc].h[1] = *(const v8bf*)(qrow + dc * 32 + hl * 8 + 16);
    }
  }

  v8f o[8] = {};
  float mrow[8], lrow[8];
#pragma unroll
  for (int r = 0; r < 8; ++r) { mrow[r] = -1e30f; lrow[r] = 0.0f; }

  const int krow = t >> 2, kseg = t & 3;
  const float scale = 0.088388347648318447f;   // 1/sqrt(128)
  const int kend = (qt + 1) * 128;

  for (int kt0 = 0; kt0 < kend; kt0 += 64) {
    __syncthreads();   // previous iteration's sK/sVT reads complete

    // K tile: one TDM DMA (64 rows x 128 elems, row pitch 136 via pad codes 5/3)
    if (wv == 0) {
      tdm_load_2d(Kg + (size_t)(b * S_ + kt0) * KVW_ + kvo,
                  (unsigned)(size_t)&sK[0][0], 128, 64, KVW_, 5, 3);
    }
    // V tile: load row-major, store transposed into LDS
    {
      const bf16* vrow_g = Vg + (size_t)(b * S_ + kt0 + krow) * KVW_ + kvo + kseg * 32;
      if (kt0 + 64 < kend)
        __builtin_prefetch(vrow_g + (size_t)64 * KVW_, 0, 1);
#pragma unroll
      for (int u = 0; u < 4; ++u) {
        v8bf vvv = *(const v8bf*)(vrow_g + u * 8);
#pragma unroll
        for (int e = 0; e < 8; ++e)
          sVT[kseg * 32 + u * 8 + e][krow] = vvv[e];
      }
    }
    if (wv == 0) __builtin_amdgcn_s_wait_tensorcnt(0);
    __syncthreads();

    // S = Q * K^T : 4 col-tiles x 4 d-chunks = 16 WMMA
    v8f sc[4] = {};
#pragma unroll
    for (int ct = 0; ct < 4; ++ct) {
#pragma unroll
      for (int dc = 0; dc < 4; ++dc) {
        FragAB kb;
        kb.h[0] = *(const v8bf*)&sK[ct * 16 + lm][dc * 32 + hl * 8];
        kb.h[1] = *(const v8bf*)&sK[ct * 16 + lm][dc * 32 + hl * 8 + 16];
        sc[ct] = wmma_bf16f32(qf[dc].v, kb.v, sc[ct]);
      }
    }

    // online softmax (fp32)
    float tm[8];
#pragma unroll
    for (int r = 0; r < 8; ++r) tm[r] = -INFINITY;
#pragma unroll
    for (int ct = 0; ct < 4; ++ct)
#pragma unroll
      for (int r = 0; r < 8; ++r) {
        float v = sc[ct][r] * scale;
        int col = kt0 + ct * 16 + lm;
        int rq  = q0 + hl * 8 + r;
        if (col > rq) v = -INFINITY;          // causal mask
        sc[ct][r] = v;
        tm[r] = fmaxf(tm[r], v);
      }
#pragma unroll
    for (int off = 1; off < 16; off <<= 1)
#pragma unroll
      for (int r = 0; r < 8; ++r)
        tm[r] = fmaxf(tm[r], __shfl_xor(tm[r], off, 32));

    float alpha[8], rs[8];
#pragma unroll
    for (int r = 0; r < 8; ++r) {
      float mn = fmaxf(fmaxf(mrow[r], tm[r]), -1e30f);
      alpha[r] = __expf(mrow[r] - mn);
      mrow[r]  = mn;
      rs[r]    = 0.0f;
    }
#pragma unroll
    for (int ct = 0; ct < 4; ++ct)
#pragma unroll
      for (int r = 0; r < 8; ++r) {
        float pv = __expf(sc[ct][r] - mrow[r]);
        rs[r] += pv;
        sP[wv][hl * 8 + r][ct * 16 + lm] = (bf16)pv;   // C-layout -> row major
      }
#pragma unroll
    for (int off = 1; off < 16; off <<= 1)
#pragma unroll
      for (int r = 0; r < 8; ++r)
        rs[r] += __shfl_xor(rs[r], off, 32);
#pragma unroll
    for (int r = 0; r < 8; ++r) lrow[r] = lrow[r] * alpha[r] + rs[r];
#pragma unroll
    for (int dt = 0; dt < 8; ++dt)
#pragma unroll
      for (int r = 0; r < 8; ++r) o[dt][r] *= alpha[r];

    __syncthreads();   // sP visibility (cross-lane A-layout reload)

    // O += P * V : 2 k-chunks x 8 d-tiles = 16 WMMA
#pragma unroll
    for (int kc2 = 0; kc2 < 2; ++kc2) {
      FragAB pf;
      pf.h[0] = *(const v8bf*)&sP[wv][lm][kc2 * 32 + hl * 8];
      pf.h[1] = *(const v8bf*)&sP[wv][lm][kc2 * 32 + hl * 8 + 16];
#pragma unroll
      for (int dt = 0; dt < 8; ++dt) {
        FragAB vf;
        vf.h[0] = *(const v8bf*)&sVT[dt * 16 + lm][kc2 * 32 + hl * 8];
        vf.h[1] = *(const v8bf*)&sVT[dt * 16 + lm][kc2 * 32 + hl * 8 + 16];
        o[dt] = wmma_bf16f32(pf.v, vf.v, o[dt]);
      }
    }
  }

  // normalize + store bf16 attn output [M, NH*D]
#pragma unroll
  for (int dt = 0; dt < 8; ++dt)
#pragma unroll
    for (int r = 0; r < 8; ++r) {
      int rq = q0 + hl * 8 + r;
      float val = o[dt][r] / lrow[r];
      O[(size_t)(b * S_ + rq) * H_ + h * D_ + dt * 16 + lm] = (bf16)val;
    }
}

// ------------------------------------------------------------------------ driver
extern "C" void kernel_launch(void* const* d_in, const int* in_sizes, int n_in,
                              void* d_out, int out_size, void* d_ws, size_t ws_size,
                              hipStream_t stream) {
  (void)in_sizes; (void)n_in; (void)out_size; (void)ws_size;
  const float* hs  = (const float*)d_in[0];
  const float* Wqf = (const float*)d_in[1];
  const float* Wkf = (const float*)d_in[2];
  const float* Wvf = (const float*)d_in[3];
  const float* Wof = (const float*)d_in[4];
  float* out = (float*)d_out;

  char* p = (char*)d_ws;
  bf16* hid = (bf16*)p;  p += (size_t)M_ * H_ * sizeof(bf16);
  bf16* wq  = (bf16*)p;  p += (size_t)H_ * H_ * sizeof(bf16);
  bf16* wk  = (bf16*)p;  p += (size_t)KVW_ * H_ * sizeof(bf16);
  bf16* wvw = (bf16*)p;  p += (size_t)KVW_ * H_ * sizeof(bf16);
  bf16* wo  = (bf16*)p;  p += (size_t)H_ * H_ * sizeof(bf16);
  bf16* q   = (bf16*)p;  p += (size_t)M_ * H_ * sizeof(bf16);
  bf16* k   = (bf16*)p;  p += (size_t)M_ * KVW_ * sizeof(bf16);
  bf16* v   = (bf16*)p;  p += (size_t)M_ * KVW_ * sizeof(bf16);
  bf16* att = (bf16*)p;  p += (size_t)M_ * H_ * sizeof(bf16);

  auto cvt = [&](const float* s, bf16* d, size_t n) {
    int n4 = (int)(n / 4);
    int blocks = (n4 + 255) / 256;
    if (blocks > 4096) blocks = 4096;
    cvt_f32_to_bf16<<<blocks, 256, 0, stream>>>(s, d, n4);
  };
  cvt(hs,  hid, (size_t)M_ * H_);
  cvt(Wqf, wq,  (size_t)H_ * H_);
  cvt(Wkf, wk,  (size_t)KVW_ * H_);
  cvt(Wvf, wvw, (size_t)KVW_ * H_);
  cvt(Wof, wo,  (size_t)H_ * H_);

  dim3 blk(256);
  gemm_xwT<bf16><<<dim3(H_ / 128,   M_ / 64), blk, 0, stream>>>(hid, wq,  q,   M_, H_,   H_);
  gemm_xwT<bf16><<<dim3(KVW_ / 128, M_ / 64), blk, 0, stream>>>(hid, wk,  k,   M_, KVW_, H_);
  gemm_xwT<bf16><<<dim3(KVW_ / 128, M_ / 64), blk, 0, stream>>>(hid, wvw, v,   M_, KVW_, H_);

  {
    int tot = M_ * NH_ * (D_ / 2);
    rope_inplace<<<(tot + 255) / 256, 256, 0, stream>>>(q, NH_, H_, tot);
  }
  {
    int tot = M_ * NKV_ * (D_ / 2);
    rope_inplace<<<(tot + 255) / 256, 256, 0, stream>>>(k, NKV_, KVW_, tot);
  }

  flash_attn<<<dim3(S_ / 128, NH_, B_), blk, 0, stream>>>(q, k, v, att);

  gemm_xwT<float><<<dim3(H_ / 128, M_ / 64), blk, 0, stream>>>(att, wo, out, M_, H_, H_);
}
